// MotionRefinement_4698694222297
// MI455X (gfx1250) — compile-verified, matrix operand order checked by reading, not run on hardware
//
#include <hip/hip_runtime.h>
#include <hip/hip_bf16.h>

#define T_REAL 2056
#define TPAD   2080            // 65*32, 130*16
#define NT16   (TPAD/16)       // 130
#define KB_PV  (TPAD/32)       // 65
#define HIDD   512
#define FFD    2048
#define NHEAD  8
#define LAYERS 6
#define NMAP   520             // rpe matrices are [520,520]

typedef _Float16 v16h __attribute__((ext_vector_type(16)));
typedef _Float16 v8h  __attribute__((ext_vector_type(8)));
typedef float    v8f  __attribute__((ext_vector_type(8)));

union V16 { v16h v; v8h p[2]; _Float16 e[16]; };

__device__ inline v8f wmma16(v16h a, v16h b, v8f c) {
  return __builtin_amdgcn_wmma_f32_16x16x32_f16(false, a, false, b, (short)0, c, false, false);
}

__device__ inline int map4(int tok) { return tok < 8 ? tok : 8 + ((tok - 8) >> 2); }

// ---- gfx1250 async global->LDS path (guarded; falls back to direct loads) ----
#if __has_builtin(__builtin_amdgcn_global_load_async_to_lds_b128)
#define HAS_ASYNC 1
#else
#define HAS_ASYNC 0
#endif

#if HAS_ASYNC
typedef int i32x4 __attribute__((ext_vector_type(4)));
typedef __attribute__((address_space(1))) i32x4* gv4p;   // pointer to AS1 int4
typedef __attribute__((address_space(3))) i32x4* lv4p;   // pointer to AS3 int4
#if __has_builtin(__builtin_amdgcn_s_wait_asynccnt)
#define WAIT_ASYNC(n) __builtin_amdgcn_s_wait_asynccnt(n)
#else
#define WAIT_ASYNC(n) asm volatile("s_wait_asynccnt " #n ::: "memory")
#endif
#else
#define WAIT_ASYNC(n)
#endif

// copy 16 bytes global -> LDS (async on gfx1250, sync fallback otherwise)
__device__ inline void async_cp16(const _Float16* src, _Float16* dst) {
#if HAS_ASYNC
  __builtin_amdgcn_global_load_async_to_lds_b128((gv4p)src, (lv4p)dst, 0, 0);
#else
  *(v8h*)dst = *(const v8h*)src;
#endif
}

// ---------------------------------------------------------------------------
// Pack fp32 row-major W[K,N] into f16 WMMA B-fragment order:
// P[((nt*KB + kb)*32 + lane)*16 + e], lane holds col n = nt*16 + lane%16,
// element e holds K = kb*32 + (e<8 ? (lane/16)*8+e : 16+(lane/16)*8+(e-8))
// ---------------------------------------------------------------------------
__global__ void pack_b(const float* __restrict__ W, _Float16* __restrict__ P,
                       int K, int N) {
  size_t i = (size_t)blockIdx.x * 256 + threadIdx.x;
  if (i >= (size_t)K * N) return;
  int e = (int)(i & 15);
  size_t t = i >> 4;
  int lane = (int)(t & 31); t >>= 5;
  int KB = K >> 5;
  int kb = (int)(t % KB);
  int nt = (int)(t / KB);
  int hf = lane >> 4, col = lane & 15;
  int kk = kb * 32 + (e < 8 ? hf * 8 + e : 16 + hf * 8 + (e - 8));
  int n  = nt * 16 + col;
  P[i] = (_Float16)W[(size_t)kk * N + n];
}

// ---------------------------------------------------------------------------
// WMMA GEMM, one wave per 16x64 output tile (4 accumulators, A reused 4x).
// K is compile-time so the K-loop unrolls; packed-B tiles staged into LDS
// with double-buffered async copies.
// ---------------------------------------------------------------------------
#define OUT_F16_ROW   0
#define OUT_F16_LEAKY 1
#define OUT_F16_TRANS 2   // write Y^T into [N][TPAD] (for V)
#define OUT_F32_RESID 3   // write resid + Y as fp32

template<int MODE, int K>
__global__ __launch_bounds__(32)
void gemm_wmma(const _Float16* __restrict__ A, const _Float16* __restrict__ Bp,
               const float* __restrict__ bias, const float* __restrict__ resid,
               _Float16* __restrict__ outH, float* __restrict__ outF, int N) {
  constexpr int KB = K / 32;
  const int lane = threadIdx.x & 31;
  const int col = lane & 15, hf = lane >> 4;
  const int nt0 = blockIdx.x * 4, mt = blockIdx.y;
  const _Float16* arow = A + (size_t)(mt * 16 + col) * K;

  __shared__ __align__(16) _Float16 bstage[2][4][512];  // 2 x 4KB

  auto issue_stage = [&](int buf, int kb) {
    for (int t = 0; t < 4; ++t) {
      const _Float16* src = Bp + (((size_t)(nt0 + t) * KB + kb) * 512) + lane * 16;
      _Float16* dst = &bstage[buf][t][lane * 16];
      async_cp16(src, dst);
      async_cp16(src + 8, dst + 8);
    }
  };

  v8f acc[4] = {{}, {}, {}, {}};
  issue_stage(0, 0);
#pragma unroll 4
  for (int kb = 0; kb < KB; ++kb) {
    const int buf = kb & 1;
    if (kb + 1 < KB) {
      issue_stage(buf ^ 1, kb + 1);
      WAIT_ASYNC(8);            // 8 outstanding = next stage; current stage done
    } else {
      WAIT_ASYNC(0);
    }
    V16 a;
    a.p[0] = *(const v8h*)(arow + kb * 32 + hf * 8);
    a.p[1] = *(const v8h*)(arow + kb * 32 + 16 + hf * 8);
#pragma unroll
    for (int t = 0; t < 4; ++t) {
      V16 b;
      b.v = *(const v16h*)&bstage[buf][t][lane * 16];
      acc[t] = wmma16(a.v, b.v, acc[t]);
    }
  }

#pragma unroll
  for (int t = 0; t < 4; ++t) {
    const int n = (nt0 + t) * 16 + col;
    const float bv = bias[n];
    if (MODE == OUT_F16_TRANS) {
      v8h pk;
      for (int r = 0; r < 8; ++r) pk[r] = (_Float16)(acc[t][r] + bv);
      *(v8h*)(outH + (size_t)n * TPAD + mt * 16 + 8 * hf) = pk;
    } else {
      for (int r = 0; r < 8; ++r) {
        int m = mt * 16 + r + 8 * hf;
        float y = acc[t][r] + bv;
        if (MODE == OUT_F16_LEAKY) y = y > 0.f ? y : 0.01f * y;
        if (MODE == OUT_F32_RESID)
          outF[(size_t)m * N + n] = resid[(size_t)m * N + n] + y;
        else
          outH[(size_t)m * N + n] = (_Float16)y;
      }
    }
  }
}

// ---------------------------------------------------------------------------
// Attention: one workgroup = (head, 16-query tile).
// Dynamic LDS: S fp32 [16][TPAD] (scores) + P f16 [16][TPAD] (probabilities).
// Phase 1: scores via WMMA + on-the-fly rel-pos bias gather.
// Phase 2: exact softmax, P written as f16.
// Phase 3: P@V with WMMA, K split across wave-pairs, combined in LDS.
// ---------------------------------------------------------------------------
__global__ __launch_bounds__(256)
void attn_kernel(const _Float16* __restrict__ q, const _Float16* __restrict__ k,
                 const _Float16* __restrict__ vT,
                 const int* __restrict__ rpe_a, const int* __restrict__ rpe_d,
                 const float* __restrict__ mask,
                 const float* __restrict__ emb_a, const float* __restrict__ emb_d,
                 _Float16* __restrict__ attout) {
  extern __shared__ float S[];                       // [16][TPAD] fp32
  _Float16* P = (_Float16*)(S + 16 * TPAD);          // [16][TPAD] f16
  __shared__ float rowscale[16];
  __shared__ float pvpart[4][32][8];                 // partial P@V from wave-group 1
  const int qt = blockIdx.x, hd = blockIdx.y;
  const int tid = threadIdx.x, w = tid >> 5, lane = tid & 31;
  const int col = lane & 15, hf = lane >> 4;

  // ---- phase 1: S = (q@k^T + pos)*rsqrt(d) + mask ----
  V16 aq[2];
  {
    const _Float16* qrow = q + (size_t)(qt * 16 + col) * HIDD + hd * 64;
    for (int kb = 0; kb < 2; ++kb) {
      aq[kb].p[0] = *(const v8h*)(qrow + kb * 32 + hf * 8);
      aq[kb].p[1] = *(const v8h*)(qrow + kb * 32 + 16 + hf * 8);
    }
  }
  for (int kt = w; kt < NT16; kt += 8) {
    v8f c = {};
    const _Float16* krow = k + (size_t)(kt * 16 + col) * HIDD + hd * 64;
#pragma unroll
    for (int kb = 0; kb < 2; ++kb) {
      V16 b;
      b.p[0] = *(const v8h*)(krow + kb * 32 + hf * 8);
      b.p[1] = *(const v8h*)(krow + kb * 32 + 16 + hf * 8);
      c = wmma16(aq[kb].v, b.v, c);
    }
    const int ktok = kt * 16 + col;
    const int km = map4(ktok < T_REAL ? ktok : 0);
    for (int r = 0; r < 8; ++r) {
      int qtok = qt * 16 + r + 8 * hf;
      float s;
      if (ktok >= T_REAL) {
        s = -1e30f;
      } else {
        int qm = map4(qtok < T_REAL ? qtok : 0);
        int idx = qm * NMAP + km;
        float pos = emb_a[rpe_a[idx] * NHEAD + hd] + emb_d[rpe_d[idx] * NHEAD + hd];
        s = (c[r] + pos) * 0.125f + mask[idx];
      }
      S[(r + 8 * hf) * TPAD + kt * 16 + col] = s;
    }
  }
  __syncthreads();

  // ---- phase 2: softmax per row (16 threads per row), write P as f16 ----
  {
    const int row = tid >> 4, sub = tid & 15;
    float* Srow = S + row * TPAD;
    _Float16* Prow = P + row * TPAD;
    float mx = -1e30f;
    for (int j = sub; j < TPAD; j += 16) mx = fmaxf(mx, Srow[j]);
    for (int off = 8; off; off >>= 1) mx = fmaxf(mx, __shfl_xor(mx, off, 16));
    float sum = 0.f;
    for (int j = sub; j < TPAD; j += 16) {
      float e = __expf(Srow[j] - mx);
      Prow[j] = (_Float16)e;
      sum += e;
    }
    for (int off = 8; off; off >>= 1) sum += __shfl_xor(sum, off, 16);
    if (sub == 0) rowscale[row] = 1.0f / sum;
  }
  __syncthreads();

  // ---- phase 3: O = (P @ V) * rowscale ; 8 waves = 4 dim-tiles x 2 K-halves ----
  {
    const int nt = w & 3, g = w >> 2;
    const int kb0 = g ? 33 : 0, kb1 = g ? KB_PV : 33;
    v8f c = {};
    const _Float16* vrow = vT + (size_t)(hd * 64 + nt * 16 + col) * TPAD;
    const _Float16* prow = P + (size_t)col * TPAD;
    for (int kb = kb0; kb < kb1; ++kb) {
      V16 a, b;
      a.p[0] = *(const v8h*)(prow + kb * 32 + hf * 8);
      a.p[1] = *(const v8h*)(prow + kb * 32 + 16 + hf * 8);
      b.p[0] = *(const v8h*)(vrow + kb * 32 + hf * 8);
      b.p[1] = *(const v8h*)(vrow + kb * 32 + 16 + hf * 8);
      c = wmma16(a.v, b.v, c);
    }
    if (g == 1)
      for (int r = 0; r < 8; ++r) pvpart[nt][lane][r] = c[r];
    __syncthreads();
    if (g == 0) {
      for (int r = 0; r < 8; ++r) {
        int qtok = qt * 16 + r + 8 * hf;
        float y = c[r] + pvpart[nt][lane][r];
        attout[(size_t)qtok * HIDD + hd * 64 + nt * 16 + col] =
            (_Float16)(y * rowscale[r + 8 * hf]);
      }
    }
  }
}

// ---------------------------------------------------------------------------
// LayerNorm per token; optional leaky; dual g/b sets (split for embedding LN).
// Writes f32 and f16 copies; zeroes padded rows.
// ---------------------------------------------------------------------------
__global__ __launch_bounds__(256)
void ln_kernel(const float* __restrict__ in,
               const float* __restrict__ g0, const float* __restrict__ b0,
               const float* __restrict__ g1, const float* __restrict__ b1,
               int split, int do_leaky,
               float* __restrict__ outF, _Float16* __restrict__ outH) {
  const int t = blockIdx.x, tid = threadIdx.x;
  if (t >= T_REAL) {
    for (int i = tid; i < HIDD; i += 256) {
      outF[(size_t)t * HIDD + i] = 0.f;
      outH[(size_t)t * HIDD + i] = (_Float16)0.f;
    }
    return;
  }
  const float* row = in + (size_t)t * HIDD;
  float v0 = row[tid], v1 = row[tid + 256];
  float s = v0 + v1, ss = v0 * v0 + v1 * v1;
  for (int off = 16; off; off >>= 1) {
    s  += __shfl_xor(s, off, 32);
    ss += __shfl_xor(ss, off, 32);
  }
  __shared__ float sh_s[8], sh_ss[8];
  if ((tid & 31) == 0) { sh_s[tid >> 5] = s; sh_ss[tid >> 5] = ss; }
  __syncthreads();
  if (tid == 0) {
    float a = 0.f, b = 0.f;
    for (int i = 0; i < 8; ++i) { a += sh_s[i]; b += sh_ss[i]; }
    sh_s[0] = a; sh_ss[0] = b;
  }
  __syncthreads();
  const float mean = sh_s[0] * (1.f / HIDD);
  const float var  = sh_ss[0] * (1.f / HIDD) - mean * mean;
  const float inv  = rsqrtf(var + 1e-5f);
  const float* g = (t < split) ? g0 : g1;
  const float* b = (t < split) ? b0 : b1;
  for (int kk = 0; kk < 2; ++kk) {
    int i = tid + kk * 256;
    float y = (row[i] - mean) * inv * g[i] + b[i];
    if (do_leaky) y = y > 0.f ? y : 0.01f * y;
    outF[(size_t)t * HIDD + i] = y;
    outH[(size_t)t * HIDD + i] = (_Float16)y;
  }
}

// ---------------------------------------------------------------------------
// Embedding pre-LN projections (tiny, VALU is fine)
// ---------------------------------------------------------------------------
__global__ void embed_obj(const float* __restrict__ obj, const float* __restrict__ Wpg,
                          const float* __restrict__ bpg, float* __restrict__ pre) {
  const int t = blockIdx.x;  // 0..7
  for (int n = threadIdx.x; n < HIDD; n += blockDim.x) {
    float acc = bpg[n];
    for (int j = 0; j < 100; ++j) acc += obj[t * 100 + j] * Wpg[j * HIDD + n];
    pre[(size_t)t * HIDD + n] = acc;
  }
}

__global__ void embed_st(const float* __restrict__ st, const float* __restrict__ Wps,
                         const float* __restrict__ bps, float* __restrict__ pre) {
  const int s = blockIdx.x;  // 0..2047
  const int stroke = s >> 2, p = s & 3;
  const float a0 = st[stroke * 10 + p * 2];
  const float a1 = st[stroke * 10 + p * 2 + 1];
  for (int n = threadIdx.x; n < HIDD; n += blockDim.x) {
    pre[(size_t)(8 + s) * HIDD + n] = a0 * Wps[n] + a1 * Wps[HIDD + n] + bps[n];
  }
}

__global__ void copy_out(const float* __restrict__ x, float* __restrict__ out, int n) {
  int i = blockIdx.x * 256 + threadIdx.x;
  if (i < n) out[i] = x[i];
}

// ---------------------------------------------------------------------------
extern "C" void kernel_launch(void* const* d_in, const int* in_sizes, int n_in,
                              void* d_out, int out_size, void* d_ws, size_t ws_size,
                              hipStream_t stream) {
  const float* obj_input = (const float*)d_in[0];
  const float* st_input0 = (const float*)d_in[2];
  const int*   rpe_a     = (const int*)d_in[3];
  const int*   rpe_d     = (const int*)d_in[4];
  const float* maskp     = (const float*)d_in[5];
  const float* Wpg = (const float*)d_in[6];
  const float* bpg = (const float*)d_in[7];
  const float* pg_g = (const float*)d_in[8];
  const float* pg_b = (const float*)d_in[9];
  const float* Wps = (const float*)d_in[10];
  const float* bps = (const float*)d_in[11];
  const float* ps_g = (const float*)d_in[12];
  const float* ps_b = (const float*)d_in[13];
  const float* emb_a = (const float*)d_in[14];
  const float* emb_d = (const float*)d_in[15];
  const float* Wq = (const float*)d_in[16];
  const float* bq = (const float*)d_in[17];
  const float* Wk = (const float*)d_in[18];
  const float* bk = (const float*)d_in[19];
  const float* Wv = (const float*)d_in[20];
  const float* bv = (const float*)d_in[21];
  const float* Wo = (const float*)d_in[22];
  const float* bo = (const float*)d_in[23];
  const float* W1 = (const float*)d_in[24];
  const float* b1 = (const float*)d_in[25];
  const float* W2 = (const float*)d_in[26];
  const float* b2 = (const float*)d_in[27];
  const float* ln1_g = (const float*)d_in[28];
  const float* ln1_b = (const float*)d_in[29];
  const float* ln2_g = (const float*)d_in[30];
  const float* ln2_b = (const float*)d_in[31];

  // ---- workspace carve-up ----
  char* wsp = (char*)d_ws;
  auto carve = [&](size_t bytes) {
    char* p = wsp;
    wsp += (bytes + 255) & ~(size_t)255;
    return p;
  };
  float*     x_f32 = (float*)carve((size_t)TPAD * HIDD * 4);
  float*     preLN = (float*)carve((size_t)TPAD * HIDD * 4);
  _Float16*  x_h   = (_Float16*)carve((size_t)TPAD * HIDD * 2);
  _Float16*  q_h   = (_Float16*)carve((size_t)TPAD * HIDD * 2);
  _Float16*  k_h   = (_Float16*)carve((size_t)TPAD * HIDD * 2);
  _Float16*  vT_h  = (_Float16*)carve((size_t)HIDD * TPAD * 2);
  _Float16*  att_h = (_Float16*)carve((size_t)TPAD * HIDD * 2);
  _Float16*  h_h   = (_Float16*)carve((size_t)TPAD * FFD * 2);
  const size_t SQ = (size_t)HIDD * HIDD;        // 262144
  const size_t SF = (size_t)HIDD * FFD;         // 1048576
  const size_t LSZ = 4 * SQ + 2 * SF;           // packed f16 elems per layer
  _Float16* pW = (_Float16*)carve(LAYERS * LSZ * 2);

  // ---- pack all weights to f16 B-fragment layout ----
  for (int l = 0; l < LAYERS; ++l) {
    _Float16* base = pW + (size_t)l * LSZ;
    pack_b<<<(unsigned)((SQ + 255) / 256), 256, 0, stream>>>(Wq + l * SQ, base,          HIDD, HIDD);
    pack_b<<<(unsigned)((SQ + 255) / 256), 256, 0, stream>>>(Wk + l * SQ, base + SQ,     HIDD, HIDD);
    pack_b<<<(unsigned)((SQ + 255) / 256), 256, 0, stream>>>(Wv + l * SQ, base + 2 * SQ, HIDD, HIDD);
    pack_b<<<(unsigned)((SQ + 255) / 256), 256, 0, stream>>>(Wo + l * SQ, base + 3 * SQ, HIDD, HIDD);
    pack_b<<<(unsigned)((SF + 255) / 256), 256, 0, stream>>>(W1 + l * SF, base + 4 * SQ,      HIDD, FFD);
    pack_b<<<(unsigned)((SF + 255) / 256), 256, 0, stream>>>(W2 + l * SF, base + 4 * SQ + SF, FFD, HIDD);
  }

  // ---- embeddings -> x ----
  embed_obj<<<8, 256, 0, stream>>>(obj_input, Wpg, bpg, preLN);
  embed_st<<<2048, 256, 0, stream>>>(st_input0, Wps, bps, preLN);
  ln_kernel<<<TPAD, 256, 0, stream>>>(preLN, pg_g, pg_b, ps_g, ps_b, 8, 1, x_f32, x_h);

  const size_t attn_lds = (size_t)16 * TPAD * 4 + (size_t)16 * TPAD * 2;  // S + P

  for (int l = 0; l < LAYERS; ++l) {
    _Float16* base = pW + (size_t)l * LSZ;
    // QKV projections
    gemm_wmma<OUT_F16_ROW, HIDD><<<dim3(HIDD / 64, NT16), 32, 0, stream>>>(
        x_h, base,          bq + l * HIDD, nullptr, q_h, nullptr, HIDD);
    gemm_wmma<OUT_F16_ROW, HIDD><<<dim3(HIDD / 64, NT16), 32, 0, stream>>>(
        x_h, base + SQ,     bk + l * HIDD, nullptr, k_h, nullptr, HIDD);
    gemm_wmma<OUT_F16_TRANS, HIDD><<<dim3(HIDD / 64, NT16), 32, 0, stream>>>(
        x_h, base + 2 * SQ, bv + l * HIDD, nullptr, vT_h, nullptr, HIDD);
    // attention
    attn_kernel<<<dim3(NT16, NHEAD), 256, attn_lds, stream>>>(
        q_h, k_h, vT_h, rpe_a, rpe_d, maskp, emb_a, emb_d, att_h);
    // O projection + residual
    gemm_wmma<OUT_F32_RESID, HIDD><<<dim3(HIDD / 64, NT16), 32, 0, stream>>>(
        att_h, base + 3 * SQ, bo + l * HIDD, x_f32, nullptr, preLN, HIDD);
    ln_kernel<<<TPAD, 256, 0, stream>>>(preLN, ln1_g + l * HIDD, ln1_b + l * HIDD,
                                        ln1_g + l * HIDD, ln1_b + l * HIDD, 0, 0, x_f32, x_h);
    // FFN
    gemm_wmma<OUT_F16_LEAKY, HIDD><<<dim3(FFD / 64, NT16), 32, 0, stream>>>(
        x_h, base + 4 * SQ, b1 + l * FFD, nullptr, h_h, nullptr, FFD);
    gemm_wmma<OUT_F32_RESID, FFD><<<dim3(HIDD / 64, NT16), 32, 0, stream>>>(
        h_h, base + 4 * SQ + SF, b2 + l * HIDD, x_f32, nullptr, preLN, HIDD);
    ln_kernel<<<TPAD, 256, 0, stream>>>(preLN, ln2_g + l * HIDD, ln2_b + l * HIDD,
                                        ln2_g + l * HIDD, ln2_b + l * HIDD, 0, 0, x_f32, x_h);
  }

  // x rows are contiguous [TPAD,512]; first 2056*512 floats are the output
  copy_out<<<(out_size + 255) / 256, 256, 0, stream>>>(x_f32, (float*)d_out, out_size);
}
